// BiLSTMCRFModule_33440615366806
// MI455X (gfx1250) — matrix operand order repaired and checked by reading, not compiled, not run on hardware
//
#include <hip/hip_runtime.h>
#include <hip/hip_bf16.h>
#include <math.h>

// Problem dims (from reference)
#define B_   32
#define S_   256
#define E_   300
#define H_   256
#define L_   17
#define G4H  1024          // 4*H
#define ROWS (B_*S_)       // 8192

typedef __bf16 bf16;
typedef bf16  v16bf __attribute__((ext_vector_type(16)));
typedef float v8f   __attribute__((ext_vector_type(8)));

__device__ __forceinline__ float sigmoidf_(float x) {
    return 1.0f / (1.0f + __expf(-x));
}
// robust fast tanh: 1 - 2/(1+e^{2x}) -> v_exp_f32 + rcp; correct limits at +-inf
__device__ __forceinline__ float tanhf_(float x) {
    return 1.0f - 2.0f / (1.0f + __expf(2.0f * x));
}

// CDNA5 async copy: 16 bytes/lane global -> LDS, tracked by ASYNCcnt.
// ldsOff is a byte offset within the workgroup LDS allocation.
__device__ __forceinline__ void async_copy16(unsigned ldsOff, const void* gptr) {
    unsigned long long ga = (unsigned long long)(uintptr_t)gptr;
    asm volatile("global_load_async_to_lds_b128 %0, %1, off"
                 :: "v"(ldsOff), "v"(ga) : "memory");
}
__device__ __forceinline__ void wait_asynccnt0() {
    asm volatile("s_wait_asynccnt 0x0" ::: "memory");
}

// ---------------------------------------------------------------------------
// Embedding gather -> bf16 padded rows (Kpad = 320 for layer 0)
// ---------------------------------------------------------------------------
__global__ void embed_to_bf16_kernel(const int* __restrict__ ids,
                                     const float* __restrict__ emb,
                                     bf16* __restrict__ X, int Kpad)
{
    int r = blockIdx.x;                 // row = b*S + t
    int w = ids[r];
    for (int k = threadIdx.x; k < Kpad; k += blockDim.x)
        X[(size_t)r * Kpad + k] = (k < E_) ? (bf16)emb[(size_t)w * E_ + k] : (bf16)0.0f;
}

// ---------------------------------------------------------------------------
// Generic f32 -> bf16 with column padding (weights, inter-layer activations)
// ---------------------------------------------------------------------------
__global__ void f32_to_bf16_kernel(const float* __restrict__ src,
                                   bf16* __restrict__ dst,
                                   int rows, int cols, int Kpad)
{
    int idx = blockIdx.x * blockDim.x + threadIdx.x;
    int total = rows * Kpad;
    if (idx >= total) return;
    int r = idx / Kpad, k = idx - r * Kpad;
    dst[idx] = (k < cols) ? (bf16)src[(size_t)r * cols + k] : (bf16)0.0f;
}

// ---------------------------------------------------------------------------
// Big input GEMM:  G = A(8192,Kpad) * W(1024,Kpad)^T + (bih+bhh)
// bf16 WMMA; wave computes a 16x64 strip. Output stored T-MAJOR:
// Gpre[(t*32 + b) * 1024 + n], so each timestep's 32x1024 block is one
// contiguous 128KB chunk (async-copy friendly for the recurrent kernel).
// ---------------------------------------------------------------------------
__global__ __launch_bounds__(256)
void gemm_xw_kernel(const bf16* __restrict__ A, const bf16* __restrict__ W,
                    const float* __restrict__ bih, const float* __restrict__ bhh,
                    float* __restrict__ G, int Kpad)
{
    const int lane = threadIdx.x & 31;
    const int wid  = blockIdx.x * (blockDim.x >> 5) + (threadIdx.x >> 5);
    const int m0   = (wid >> 4) << 4;      // 512 row tiles
    const int n0   = (wid & 15) << 6;      // 16 col groups of 64
    const int mA   = m0 + (lane & 15);
    const int aSel = (lane >> 4) ? 8  : 0; // A: hi lanes hold K+8..15 / K+24..31
    const int bSel = (lane >> 4) ? 8  : 0; // B: hi lanes hold K+8..15 / K+24..31

    v8f acc[4] = {};

    for (int k0 = 0; k0 < Kpad; k0 += 32) {
        v16bf a;
        const bf16* pa = A + (size_t)mA * Kpad + k0 + aSel;
        #pragma unroll
        for (int i = 0; i < 8; ++i) { a[i] = pa[i]; a[8 + i] = pa[16 + i]; }
        #pragma unroll
        for (int j = 0; j < 4; ++j) {
            const bf16* pb = W + (size_t)(n0 + j * 16 + (lane & 15)) * Kpad + k0 + bSel;
            v16bf b;
            #pragma unroll
            for (int i = 0; i < 8; ++i) { b[i] = pb[i]; b[8 + i] = pb[16 + i]; }
            acc[j] = __builtin_amdgcn_wmma_f32_16x16x32_bf16(
                false, a, false, b, (short)0, acc[j], false, false);
        }
    }

    const int mb = m0 + aSel;
    #pragma unroll
    for (int j = 0; j < 4; ++j) {
        int n = n0 + j * 16 + (lane & 15);
        float bias = bih[n] + bhh[n];
        #pragma unroll
        for (int i = 0; i < 8; ++i) {
            int m = mb + i;                       // m = b*256 + t
            int b = m >> 8, t = m & (S_ - 1);
            G[((size_t)t * B_ + b) * G4H + n] = acc[j][i] + bias;
        }
    }
}

// ---------------------------------------------------------------------------
// Recurrent LSTM layer: one persistent 512-thread workgroup per direction.
// Per step t:  g = Gpre[t] (LDS, async-prefetched) + h_{t-1} @ Whh^T (WMMA)
// LDS map (dynamic, 304KB):
//   [0,16K)    h  bf16[32][256]   (A operand source)
//   [16K,48K)  c  f32 [32][256]
//   [48K,176K) Gpre ping buffer f32[32][1024]  (reused in-place as gate buf)
//   [176K,304K) Gpre pong buffer
// 16 waves; each wave owns a 64-col group x both 16-row batch tiles
// (B fragment loaded once, reused for both M tiles).
// ---------------------------------------------------------------------------
#define GBUF0 (48u * 1024u)
#define GBUF1 (176u * 1024u)

__global__ __launch_bounds__(512)
void lstm_layer_kernel(const float* __restrict__ Gpre0, const float* __restrict__ Gpre1,
                       const bf16* __restrict__ Whh0,  const bf16* __restrict__ Whh1,
                       float* __restrict__ feats)
{
    extern __shared__ char smem_raw[];
    bf16*  h_sh = (bf16*)smem_raw;                    // [32][256]
    float* c_sh = (float*)(smem_raw + 16 * 1024);     // [32][256]

    const int dir = blockIdx.x;
    const float* Gpre = dir ? Gpre1 : Gpre0;
    const bf16*  Whh  = dir ? Whh1  : Whh0;

    const int tid  = threadIdx.x;
    const int lane = tid & 31;
    const int wid  = tid >> 5;                 // 0..15
    const int n0   = wid << 6;                 // 64-wide gate-column group
    const int aSel = (lane >> 4) ? 8  : 0;
    const int bSel = (lane >> 4) ? 8  : 0;

    // Prologue: async-stage Gpre[t_first] into ping buffer
    {
        const int t0 = dir ? (S_ - 1) : 0;
        const char* gb = (const char*)(Gpre + (size_t)t0 * (B_ * G4H));
        #pragma unroll
        for (int r = 0; r < 16; ++r)
            async_copy16(GBUF0 + tid * 16 + r * 8192, gb + tid * 16 + r * 8192);
    }
    for (int i = tid; i < B_ * H_; i += 512) { h_sh[i] = (bf16)0.0f; c_sh[i] = 0.0f; }
    wait_asynccnt0();
    __syncthreads();

    unsigned curOff = GBUF0;
    for (int s = 0; s < S_; ++s) {
        const int t = dir ? (S_ - 1 - s) : s;
        const unsigned nxtOff = (curOff == GBUF0) ? GBUF1 : GBUF0;

        // Kick off async prefetch of next step's Gpre tile (hidden behind WMMA)
        if (s + 1 < S_) {
            const int tn = dir ? (S_ - 2 - s) : (s + 1);
            const char* gb = (const char*)(Gpre + (size_t)tn * (B_ * G4H));
            #pragma unroll
            for (int r = 0; r < 16; ++r)
                async_copy16(nxtOff + tid * 16 + r * 8192, gb + tid * 16 + r * 8192);
        }

        float* Gcur = (float*)(smem_raw + curOff);    // [32][1024]

        // acc <- staged Gpre tile (each wave reads exactly the cells it will
        // later overwrite with gate pre-activations -> safe in-place reuse)
        v8f acc[2][4];
        #pragma unroll
        for (int mt = 0; mt < 2; ++mt) {
            const int mb = (mt << 4) + aSel;
            #pragma unroll
            for (int j = 0; j < 4; ++j) {
                int n = n0 + j * 16 + (lane & 15);
                #pragma unroll
                for (int i = 0; i < 8; ++i)
                    acc[mt][j][i] = Gcur[(mb + i) * G4H + n];
            }
        }

        // acc += h_{t-1} @ Whh^T : K=256, B fragment reused across both M tiles
        for (int k0 = 0; k0 < H_; k0 += 32) {
            v16bf a0, a1;
            const bf16* pa0 = h_sh + (0  + (lane & 15)) * H_ + k0 + aSel;
            const bf16* pa1 = h_sh + (16 + (lane & 15)) * H_ + k0 + aSel;
            #pragma unroll
            for (int i = 0; i < 8; ++i) {
                a0[i] = pa0[i]; a0[8 + i] = pa0[16 + i];
                a1[i] = pa1[i]; a1[8 + i] = pa1[16 + i];
            }
            #pragma unroll
            for (int j = 0; j < 4; ++j) {
                const bf16* pb = Whh + (size_t)(n0 + j * 16 + (lane & 15)) * H_ + k0 + bSel;
                v16bf b;
                #pragma unroll
                for (int i = 0; i < 8; ++i) { b[i] = pb[i]; b[8 + i] = pb[16 + i]; }
                acc[0][j] = __builtin_amdgcn_wmma_f32_16x16x32_bf16(
                    false, a0, false, b, (short)0, acc[0][j], false, false);
                acc[1][j] = __builtin_amdgcn_wmma_f32_16x16x32_bf16(
                    false, a1, false, b, (short)0, acc[1][j], false, false);
            }
        }

        // stage gate pre-activations in-place over the consumed Gpre tile
        #pragma unroll
        for (int mt = 0; mt < 2; ++mt) {
            const int mb = (mt << 4) + aSel;
            #pragma unroll
            for (int j = 0; j < 4; ++j) {
                int n = n0 + j * 16 + (lane & 15);
                #pragma unroll
                for (int i = 0; i < 8; ++i)
                    Gcur[(mb + i) * G4H + n] = acc[mt][j][i];
            }
        }
        __syncthreads();

        // gate nonlinearity + state update: 16 (b,h) cells per thread
        #pragma unroll
        for (int kk = 0; kk < 16; ++kk) {
            int cell = tid + kk * 512;           // 0..8191
            int b  = cell >> 8;
            int hh = cell & 255;
            const float* gb = Gcur + b * G4H;
            float ig = sigmoidf_(gb[hh]);
            float fg = sigmoidf_(gb[H_ + hh]);
            float gg = tanhf_  (gb[2 * H_ + hh]);
            float og = sigmoidf_(gb[3 * H_ + hh]);
            float c  = fg * c_sh[cell] + ig * gg;
            c_sh[cell] = c;
            float h = og * tanhf_(c);
            h_sh[cell] = (bf16)h;
            feats[((size_t)b * S_ + t) * (2 * H_) + dir * H_ + hh] = h;
        }
        wait_asynccnt0();     // next tile fully in LDS (own wave's ops)...
        __syncthreads();      // ...and everyone else's too
        curOff = nxtOff;
    }
}

// ---------------------------------------------------------------------------
// Emission projection: em(8192,17) = feats(8192,512) @ out_w(17,512)^T + out_b
// ---------------------------------------------------------------------------
__global__ __launch_bounds__(32)
void proj_kernel(const float* __restrict__ feats, const float* __restrict__ out_w,
                 const float* __restrict__ out_b, float* __restrict__ em)
{
    int r = blockIdx.x;
    int l = threadIdx.x;
    if (l >= L_) return;
    const float* f = feats + (size_t)r * (2 * H_);
    const float* w = out_w + (size_t)l * (2 * H_);
    float sum = out_b[l];
    for (int k = 0; k < 2 * H_; ++k) sum += f[k] * w[k];
    em[(size_t)r * L_ + l] = sum;
}

// ---------------------------------------------------------------------------
// CRF forward (logZ) + gold score; one wave per batch element.
// ---------------------------------------------------------------------------
__global__ __launch_bounds__(32)
void crf_nll_kernel(const float* __restrict__ em, const int* __restrict__ tags,
                    const int* __restrict__ mask, const float* __restrict__ start,
                    const float* __restrict__ endv, const float* __restrict__ trans,
                    float* __restrict__ nll_b)
{
    __shared__ float alpha[L_];
    int b = blockIdx.x;
    int j = threadIdx.x;
    const float* emb = em + (size_t)b * S_ * L_;
    const int* tg = tags + b * S_;
    const int* mk = mask + b * S_;

    if (j < L_) alpha[j] = start[j] + emb[j];
    __syncthreads();

    for (int t = 1; t < S_; ++t) {
        float v = 0.0f;
        if (j < L_) {
            float mx = -INFINITY;
            for (int i = 0; i < L_; ++i)
                mx = fmaxf(mx, alpha[i] + trans[i * L_ + j]);
            float sm = 0.0f;
            for (int i = 0; i < L_; ++i)
                sm += __expf(alpha[i] + trans[i * L_ + j] - mx);
            v = mx + __logf(sm) + emb[t * L_ + j];
        }
        __syncthreads();
        if (j < L_ && mk[t]) alpha[j] = v;
        __syncthreads();
    }

    if (j == 0) {
        float mx = -INFINITY;
        for (int i = 0; i < L_; ++i) mx = fmaxf(mx, alpha[i] + endv[i]);
        float sm = 0.0f;
        for (int i = 0; i < L_; ++i) sm += __expf(alpha[i] + endv[i] - mx);
        float logZ = mx + __logf(sm);

        float score = start[tg[0]] + emb[tg[0]];
        int len = 0;
        for (int t = 0; t < S_; ++t) len += (mk[t] ? 1 : 0);
        for (int t = 1; t < S_; ++t) {
            float inc = trans[tg[t - 1] * L_ + tg[t]] + emb[t * L_ + tg[t]];
            score += mk[t] ? inc : 0.0f;
        }
        score += endv[tg[len - 1]];
        nll_b[b] = score - logZ;
    }
}

// ---------------------------------------------------------------------------
// Viterbi decode; one wave per batch element; backtrace by lane 0.
// ---------------------------------------------------------------------------
__global__ __launch_bounds__(32)
void viterbi_kernel(const float* __restrict__ em, const int* __restrict__ mask,
                    const float* __restrict__ start, const float* __restrict__ endv,
                    const float* __restrict__ trans, int* __restrict__ bp,
                    float* __restrict__ preds)
{
    __shared__ float score[L_];
    int b = blockIdx.x;
    int j = threadIdx.x;
    const float* emb = em + (size_t)b * S_ * L_;
    const int* mk = mask + b * S_;
    int* bpb = bp + (size_t)b * S_ * L_;

    if (j < L_) score[j] = start[j] + emb[j];
    __syncthreads();

    for (int t = 1; t < S_; ++t) {
        float best = -INFINITY; int arg = 0;
        if (j < L_) {
            for (int i = 0; i < L_; ++i) {
                float x = score[i] + trans[i * L_ + j];
                if (x > best) { best = x; arg = i; }
            }
            best += emb[t * L_ + j];
        }
        __syncthreads();
        if (j < L_) {
            if (mk[t]) { score[j] = best; bpb[t * L_ + j] = arg; }
            else       { bpb[t * L_ + j] = j; }
        }
        __syncthreads();
    }

    if (j == 0) {
        int tag = 0; float best = -INFINITY;
        for (int i = 0; i < L_; ++i) {
            float x = score[i] + endv[i];
            if (x > best) { best = x; tag = i; }
        }
        for (int t = S_ - 1; t >= 1; --t) {
            preds[(size_t)b * S_ + t] = mk[t] ? (float)tag : 0.0f;
            tag = bpb[t * L_ + tag];
        }
        preds[(size_t)b * S_] = mk[0] ? (float)tag : 0.0f;
    }
}

__global__ void finalize_kernel(const float* __restrict__ nll_b, float* __restrict__ out0)
{
    if (threadIdx.x == 0) {
        float s = 0.0f;
        for (int b = 0; b < B_; ++b) s += nll_b[b];
        out0[0] = -s / (float)B_;
    }
}

// ---------------------------------------------------------------------------
// Launch
// ---------------------------------------------------------------------------
extern "C" void kernel_launch(void* const* d_in, const int* in_sizes, int n_in,
                              void* d_out, int out_size, void* d_ws, size_t ws_size,
                              hipStream_t stream)
{
    (void)in_sizes; (void)n_in; (void)out_size; (void)ws_size;

    const int*   word_ids  = (const int*)d_in[0];
    const int*   mask      = (const int*)d_in[1];
    const int*   tag_ids   = (const int*)d_in[2];
    const float* word_emb  = (const float*)d_in[3];
    // d_in[4..35]: 8 x (Wih, Whh, bih, bhh) in (layer, dir) order
    const float* out_w     = (const float*)d_in[36];
    const float* out_b     = (const float*)d_in[37];
    const float* crf_start = (const float*)d_in[38];
    const float* crf_end   = (const float*)d_in[39];
    const float* crf_trans = (const float*)d_in[40];

    char* ws = (char*)d_ws;
    size_t off = 0;
    auto alloc = [&](size_t bytes) -> void* {
        void* p = ws + off;
        off = (off + bytes + 255) & ~(size_t)255;
        return p;
    };

    const int KP0 = 320;   // 300 padded to multiple of 32
    const int KP1 = 512;

    bf16*  Xbf    = (bf16*) alloc((size_t)ROWS * KP1 * sizeof(bf16));
    bf16*  Wbf    = (bf16*) alloc((size_t)G4H * KP1 * sizeof(bf16));
    bf16*  Whh0b  = (bf16*) alloc((size_t)G4H * H_ * sizeof(bf16));
    bf16*  Whh1b  = (bf16*) alloc((size_t)G4H * H_ * sizeof(bf16));
    float* Gpre0  = (float*)alloc((size_t)ROWS * G4H * sizeof(float));
    float* Gpre1  = (float*)alloc((size_t)ROWS * G4H * sizeof(float));
    float* feats0 = (float*)alloc((size_t)ROWS * 2 * H_ * sizeof(float));
    float* feats1 = (float*)alloc((size_t)ROWS * 2 * H_ * sizeof(float));
    float* emis   = (float*)alloc((size_t)ROWS * L_ * sizeof(float));
    int*   bp     = (int*)  alloc((size_t)ROWS * L_ * sizeof(int));
    float* nll_b  = (float*)alloc((size_t)B_ * sizeof(float));

    embed_to_bf16_kernel<<<ROWS, 128, 0, stream>>>(word_ids, word_emb, Xbf, KP0);

    for (int layer = 0; layer < 2; ++layer) {
        const int Kpad   = layer ? KP1 : KP0;
        const int in_dim = layer ? 2 * H_ : E_;

        if (layer) {
            int total = ROWS * KP1;
            f32_to_bf16_kernel<<<(total + 255) / 256, 256, 0, stream>>>(
                feats0, Xbf, ROWS, 2 * H_, KP1);
        }

        for (int dir = 0; dir < 2; ++dir) {
            const float* Wih = (const float*)d_in[4 + 4 * (2 * layer + dir) + 0];
            const float* Whh = (const float*)d_in[4 + 4 * (2 * layer + dir) + 1];
            const float* bih = (const float*)d_in[4 + 4 * (2 * layer + dir) + 2];
            const float* bhh = (const float*)d_in[4 + 4 * (2 * layer + dir) + 3];

            int wtot = G4H * Kpad;
            f32_to_bf16_kernel<<<(wtot + 255) / 256, 256, 0, stream>>>(
                Wih, Wbf, G4H, in_dim, Kpad);

            gemm_xw_kernel<<<1024, 256, 0, stream>>>(
                Xbf, Wbf, bih, bhh, dir ? Gpre1 : Gpre0, Kpad);

            int htot = G4H * H_;
            f32_to_bf16_kernel<<<(htot + 255) / 256, 256, 0, stream>>>(
                Whh, dir ? Whh1b : Whh0b, G4H, H_, H_);
        }

        float* fo = layer ? feats1 : feats0;
        lstm_layer_kernel<<<2, 512, 304 * 1024, stream>>>(
            Gpre0, Gpre1, Whh0b, Whh1b, fo);
    }

    proj_kernel<<<ROWS, 32, 0, stream>>>(feats1, out_w, out_b, emis);

    float* outf = (float*)d_out;
    crf_nll_kernel<<<B_, 32, 0, stream>>>(emis, tag_ids, mask,
                                          crf_start, crf_end, crf_trans, nll_b);
    viterbi_kernel<<<B_, 32, 0, stream>>>(emis, mask,
                                          crf_start, crf_end, crf_trans, bp, outf + 1);
    finalize_kernel<<<1, 32, 0, stream>>>(nll_b, outf);
}
